// GraphGatedLayerLinear_51101520888314
// MI455X (gfx1250) — compile-verified, matrix operand order checked by reading, not run on hardware
//
#include <hip/hip_runtime.h>
#include <hip/hip_bf16.h>

// ---------------------------------------------------------------------------
// GraphGatedLayerLinear for MI455X (gfx1250, wave32, WMMA)
// N=64, K=64, C=256, G=4.  M = N*K = 4096 flattened rows.
//
// Key algebraic rewrite: the masked pairwise GEMM
//   msg = (sum_i E[i,j] * (tanh(z[n,i,j]) @ W_b2 + b_b2)) / (1+deg)
// commutes with the linear map:
//   msg[n,j] = ( (sum_{i != j} tanh(zj[n,j]+zi[n,i]+b_b1)) @ W_b2 + 63*b_b2 ) / 64
// so the O(N*K*K*C*C) GEMM collapses to O(N*K*K*C) tanh-adds + one 4096x256x256 GEMM.
// Everything (~21 MB) is L2-resident on MI455X (192 MB L2); HBM is irrelevant.
// ---------------------------------------------------------------------------

typedef __attribute__((ext_vector_type(16))) _Float16 v16h;
typedef __attribute__((ext_vector_type(8)))  _Float16 v8h;
typedef __attribute__((ext_vector_type(8)))  float    v8f;

#define Cc 256
#define Kk 64
#define Nn 64
#define Mm 4096
#define Gg 4

__device__ __forceinline__ float fast_tanh(float x) {
#if __has_builtin(__builtin_amdgcn_tanhf)
    return __builtin_amdgcn_tanhf(x);            // gfx1250 v_tanh_f32
#else
    // tanh(x) = 1 - 2/(exp(2x)+1); exp2 overflow -> inf -> rcp -> 0 -> +/-1 (correct saturation)
    float e = __builtin_amdgcn_exp2f(x * 2.8853900817779268f); // 2*log2(e)
    return 1.0f - 2.0f * __builtin_amdgcn_rcpf(e + 1.0f);
#endif
}

__device__ __forceinline__ float fast_sigmoid(float x) {
    float e = __builtin_amdgcn_exp2f(-x * 1.4426950408889634f);
    return __builtin_amdgcn_rcpf(1.0f + e);
}

// --- weight convert+transpose: Wt[d][cin] = (f16) W[(row0+cin)*256 + d] -----
__global__ void wconv_kernel(const float* __restrict__ W, int row0,
                             _Float16* __restrict__ Wt) {
    int cin = blockIdx.x;        // 0..255 (input channel)
    int d   = threadIdx.x;       // 0..255 (output channel)
    Wt[d * Cc + cin] = (_Float16)W[(row0 + cin) * Cc + d];
}

// --- activation convert: xc16[m][c] = (f16) x[m*260 + c] --------------------
__global__ void xconv_kernel(const float* __restrict__ x, _Float16* __restrict__ xc16) {
    int m = blockIdx.x, c = threadIdx.x;
    xc16[m * Cc + c] = (_Float16)x[m * (Cc + Gg) + c];
}

// --- generic 4096x256x256 GEMM via v_wmma_f32_16x16x32_f16 ------------------
// Out[m][n] = act( (sum_k A[m][k]*Bt[n][k]  + biasMult*bias[n] + t*w0[n]) * scale )
// One wave -> one 16x16 tile; 8 waves / 256-thread block; 8 k32 WMMA steps.
__global__ void gemm_wmma_kernel(const _Float16* __restrict__ A,
                                 const _Float16* __restrict__ Bt,
                                 const float* __restrict__ bias, float biasMult,
                                 float scale,
                                 const float* __restrict__ w0,
                                 const float* __restrict__ tptr,
                                 float* __restrict__ outF,
                                 _Float16* __restrict__ outH,
                                 int act) {
    const int lane = threadIdx.x & 31;
    const int wid  = threadIdx.x >> 5;
    const int tile = blockIdx.x * 8 + wid;      // 4096 tiles total
    const int tn = tile & 15;                   // 16 tiles along N (=256/16)
    const int tm = tile >> 4;                   // 256 tiles along M (=4096/16)

    const int r  = lane & 15;
    const int hi = lane >> 4;                   // lane half selects K sub-offset
    const int rowA = tm * 16 + r;               // this lane's A row
    const int rowB = tn * 16 + r;               // this lane's output column (Bt row)

    const _Float16* ap = A  + rowA * Cc + hi * 8;
    const _Float16* bp = Bt + rowB * Cc + hi * 8;

    v8f acc = {};
#pragma unroll
    for (int kk = 0; kk < 8; ++kk) {
        const int kb = kk * 32;
        v8h a0 = *(const v8h*)(ap + kb);
        v8h a1 = *(const v8h*)(ap + kb + 16);
        v8h b0 = *(const v8h*)(bp + kb);
        v8h b1 = *(const v8h*)(bp + kb + 16);
        v16h a = __builtin_shufflevector(a0, a1, 0,1,2,3,4,5,6,7,8,9,10,11,12,13,14,15);
        v16h b = __builtin_shufflevector(b0, b1, 0,1,2,3,4,5,6,7,8,9,10,11,12,13,14,15);
        acc = __builtin_amdgcn_wmma_f32_16x16x32_f16(
            /*neg_a=*/false, a, /*neg_b=*/false, b,
            /*c_mod=*/(short)0, acc, /*reuse_a=*/false, /*reuse_b=*/false);
    }

    const int n = tn * 16 + r;
    float add = 0.0f;
    if (bias) add += biasMult * bias[n];
    if (w0)   add += tptr[0] * w0[n];           // rank-1 term t * W_lm[0, n]

#pragma unroll
    for (int v = 0; v < 8; ++v) {
        int m = tm * 16 + v + hi * 8;           // C/D layout: row = v + (lane>>4)*8
        float val = (acc[v] + add) * scale;
        if (act) val = fast_tanh(val);
        if (outF) outF[m * Cc + n] = val;
        if (outH) outH[m * Cc + n] = (_Float16)val;
    }
}

// --- pairwise tanh reduction: s[n,j,c] = sum_{i!=j} tanh(zj[n,j,c]+zi[n,i,c]+b_b1[c])
__global__ void pairwise_kernel(const float* __restrict__ zj,
                                const float* __restrict__ zi,
                                const float* __restrict__ b_b1,
                                _Float16* __restrict__ s16) {
    const int j = blockIdx.x, n = blockIdx.y, c = threadIdx.x;
    const float zjv = zj[(n * Kk + j) * Cc + c] + b_b1[c];
    const float* zin = zi + (n * Kk) * Cc + c;
    float acc = 0.0f;
#pragma unroll 8
    for (int i = 0; i < Kk; ++i) {
        float v = fast_tanh(zjv + zin[i * Cc]);
        acc += (i == j) ? 0.0f : v;
    }
    s16[(n * Kk + j) * Cc + c] = (_Float16)acc;   // |acc| <= 63, exact range for f16
}

// --- finalize: gate = sigmoid(bg@W_g+b_g); out = [unary + msgPre*gate + h, gate]
__global__ void finalize_kernel(const float* __restrict__ x,
                                const float* __restrict__ hF,
                                const float* __restrict__ unary,
                                const float* __restrict__ msgPre,
                                const float* __restrict__ W_g,
                                const float* __restrict__ b_g,
                                float* __restrict__ out) {
    const int m = blockIdx.x;
    __shared__ float gate_s[Gg];
    if (threadIdx.x < Gg) {
        int g = threadIdx.x;
        float acc = b_g[g];
#pragma unroll
        for (int g2 = 0; g2 < Gg; ++g2)
            acc += x[m * (Cc + Gg) + Cc + g2] * W_g[g2 * Gg + g];
        float gt = fast_sigmoid(acc);
        gate_s[g] = gt;
        out[m * (Cc + Gg) + Cc + g] = gt;
    }
    __syncthreads();
    const int c = threadIdx.x;
    out[m * (Cc + Gg) + c] =
        unary[m * Cc + c] + msgPre[m * Cc + c] * gate_s[c >> 6] + hF[m * Cc + c];
}

extern "C" void kernel_launch(void* const* d_in, const int* in_sizes, int n_in,
                              void* d_out, int out_size, void* d_ws, size_t ws_size,
                              hipStream_t stream) {
    (void)in_sizes; (void)n_in; (void)out_size; (void)ws_size;
    const float* t    = (const float*)d_in[0];
    const float* x    = (const float*)d_in[1];
    const float* W_lm = (const float*)d_in[2];
    const float* b_lm = (const float*)d_in[3];
    const float* W_u1 = (const float*)d_in[4];
    const float* b_u1 = (const float*)d_in[5];
    const float* W_u2 = (const float*)d_in[6];
    const float* b_u2 = (const float*)d_in[7];
    const float* W_b1 = (const float*)d_in[8];
    const float* b_b1 = (const float*)d_in[9];
    const float* W_b2 = (const float*)d_in[10];
    const float* b_b2 = (const float*)d_in[11];
    const float* W_g  = (const float*)d_in[12];
    const float* b_g  = (const float*)d_in[13];
    float* out = (float*)d_out;

    // ---- workspace layout (aliased where lifetimes allow) ----
    char* ws = (char*)d_ws;
    size_t off = 0;
    auto take = [&](size_t bytes) { char* p = ws + off; off += (bytes + 255) & ~size_t(255); return p; };
    float*    hF     = (float*)take(Mm * Cc * 4);   // h (f32), live to finalize
    float*    unary  = (float*)take(Mm * Cc * 4);   // unary branch out
    float*    zjF    = (float*)take(Mm * Cc * 4);   // zj; reused as msgPre after pairwise
    float*    ziF    = (float*)take(Mm * Cc * 4);   // zi
    _Float16* h16    = (_Float16*)take(Mm * Cc * 2);
    _Float16* buf16  = (_Float16*)take(Mm * Cc * 2); // xc16 -> a1_16 -> s16 (serial reuse)
    _Float16* Wlm_t  = (_Float16*)take(Cc * Cc * 2);
    _Float16* Wu1_t  = (_Float16*)take(Cc * Cc * 2);
    _Float16* Wu2_t  = (_Float16*)take(Cc * Cc * 2);
    _Float16* Wb1j_t = (_Float16*)take(Cc * Cc * 2);
    _Float16* Wb1i_t = (_Float16*)take(Cc * Cc * 2);
    _Float16* Wb2_t  = (_Float16*)take(Cc * Cc * 2);
    float*    msgPre = zjF;

    // ---- 1) weight transpose+f16 convert (tiny, L2-resident thereafter) ----
    wconv_kernel<<<Cc, Cc, 0, stream>>>(W_lm, 1, Wlm_t);    // rows 1..256 (xc part)
    wconv_kernel<<<Cc, Cc, 0, stream>>>(W_u1, 0, Wu1_t);
    wconv_kernel<<<Cc, Cc, 0, stream>>>(W_u2, 0, Wu2_t);
    wconv_kernel<<<Cc, Cc, 0, stream>>>(W_b1, 0, Wb1j_t);   // W_b1[:256]
    wconv_kernel<<<Cc, Cc, 0, stream>>>(W_b1, Cc, Wb1i_t);  // W_b1[256:]
    wconv_kernel<<<Cc, Cc, 0, stream>>>(W_b2, 0, Wb2_t);

    // ---- 2) x channels -> f16 ----
    xconv_kernel<<<Mm, Cc, 0, stream>>>(x, buf16);          // buf16 = xc16

    const dim3 gGrid(Mm * (Cc / 16) / 16 / 8);              // 4096 tiles / 8 waves = 512 blocks
    const dim3 gBlk(256);

    // ---- 3) h = xc@W_lm[1:] + t*W_lm[0] + b_lm  (f32 + f16 copies) ----
    gemm_wmma_kernel<<<gGrid, gBlk, 0, stream>>>(buf16, Wlm_t, b_lm, 1.0f, 1.0f,
                                                 W_lm /*row0*/, t, hF, h16, 0);
    // ---- 4) a1 = tanh(h@W_u1 + b_u1) (f16) ----
    gemm_wmma_kernel<<<gGrid, gBlk, 0, stream>>>(h16, Wu1_t, b_u1, 1.0f, 1.0f,
                                                 nullptr, nullptr, nullptr, buf16, 1);
    // ---- 5) unary = a1@W_u2 + b_u2 (f32) ----
    gemm_wmma_kernel<<<gGrid, gBlk, 0, stream>>>(buf16, Wu2_t, b_u2, 1.0f, 1.0f,
                                                 nullptr, nullptr, unary, nullptr, 0);
    // ---- 6) zj = h@W_b1[:256], zi = h@W_b1[256:] (f32, no bias) ----
    gemm_wmma_kernel<<<gGrid, gBlk, 0, stream>>>(h16, Wb1j_t, nullptr, 0.0f, 1.0f,
                                                 nullptr, nullptr, zjF, nullptr, 0);
    gemm_wmma_kernel<<<gGrid, gBlk, 0, stream>>>(h16, Wb1i_t, nullptr, 0.0f, 1.0f,
                                                 nullptr, nullptr, ziF, nullptr, 0);
    // ---- 7) pairwise tanh reduction (VALU hot loop, 67M tanh) ----
    pairwise_kernel<<<dim3(Kk, Nn), Cc, 0, stream>>>(zjF, ziF, b_b1, buf16); // buf16 = s16
    // ---- 8) msgPre = (s@W_b2 + 63*b_b2)/64  (overwrites zj, now dead) ----
    gemm_wmma_kernel<<<gGrid, gBlk, 0, stream>>>(buf16, Wb2_t, b_b2, 63.0f, 1.0f / 64.0f,
                                                 nullptr, nullptr, msgPre, nullptr, 0);
    // ---- 9) gate + group gating + residual + concat ----
    finalize_kernel<<<Mm, Cc, 0, stream>>>(x, hF, unary, msgPre, W_g, b_g, out);
}
